// DocModel_20693152432139
// MI455X (gfx1250) — matrix operand order repaired and checked by reading, not compiled, not run on hardware
//
#include <hip/hip_runtime.h>
#include <hip/hip_bf16.h>

typedef __attribute__((ext_vector_type(16))) _Float16 v16h;
typedef __attribute__((ext_vector_type(8)))  _Float16 v8h;
typedef __attribute__((ext_vector_type(8)))  float    v8f;

// Problem constants (from reference): B=2 D=12 P=32 S=255 E=100 U=128 H=256
#define E_DIM   100
#define NSENT   768      // B*D*P
#define SLEN    255
#define NPARA   24       // B*D
#define PLEN    32
#define NDOC    2        // B
#define DLEN    12
#define NGATE   512      // 4*U
#define NT      32       // 512/16 column tiles

__device__ __forceinline__ float sigf(float x) { return 1.0f / (1.0f + __expf(-x)); }

// ---- A fragment (16x32 f16, M x K), per ISA 7.12.2:
// lanes 0-15: M=lane, halves 0..7 -> K 0..7, halves 8..15 -> K 16..23
// lanes 16-31: M=lane-16, halves 0..7 -> K 8..15, halves 8..15 -> K 24..31
__device__ __forceinline__ v16h load_a16(const _Float16* base, size_t row_stride_h, int lane) {
    const int m  = lane & 15;
    const int kb = (lane >> 4) * 8;
    const _Float16* p = base + (size_t)m * row_stride_h + kb;
    v8h lo = *(const v8h*)(p);          // K = kb .. kb+7       (16B aligned)
    v8h hi = *(const v8h*)(p + 16);     // K = kb+16 .. kb+23
    return __builtin_shufflevector(lo, hi, 0,1,2,3,4,5,6,7,8,9,10,11,12,13,14,15);
}

// ------------------------------------------------------------------
// Embedding gather: tokens -> f16 rows padded E(100) -> 128
// ------------------------------------------------------------------
__global__ void embed_gather(const int* __restrict__ tokens,
                             const float* __restrict__ emb,
                             _Float16* __restrict__ out, int nrows) {
    int idx = blockIdx.x * blockDim.x + threadIdx.x;
    int r = idx >> 7, k = idx & 127;
    if (r >= nrows) return;
    float v = 0.0f;
    if (k < E_DIM) v = emb[(size_t)tokens[r] * E_DIM + k];
    out[idx] = (_Float16)v;
}

// ------------------------------------------------------------------
// Weight prep: concat [Wx (ka x 512, zero-padded to kxpad rows); Wh (128 x 512)]
// pre-swizzled into WMMA B-fragment order so the LSTM kernel loads one v16h/lane.
// Fragment element e of lane L of tile (kc,nt):
//   n = nt*16 + (L&15), K = kc*32 + (L>>4)*8 + (e&7) + (e>>3)*16
// ------------------------------------------------------------------
__global__ void prep_wcat(const float* __restrict__ Wx, int ka,
                          const float* __restrict__ Wh,
                          _Float16* __restrict__ wfrag, int kxpad, int nk32) {
    int idx = blockIdx.x * blockDim.x + threadIdx.x;
    int total = nk32 * NT * 32 * 16;
    if (idx >= total) return;
    int e    = idx & 15;
    int lane = (idx >> 4) & 31;
    int nt   = (idx >> 9) & 31;
    int kc   = idx >> 14;
    int n  = nt * 16 + (lane & 15);
    int kg = kc * 32 + (lane >> 4) * 8 + (e & 7) + (e >> 3) * 16;
    float v;
    if (kg < kxpad) v = (kg < ka) ? Wx[(size_t)kg * NGATE + n] : 0.0f;
    else            v = Wh[(size_t)(kg - kxpad) * NGATE + n];
    wfrag[idx] = (_Float16)v;
}

// ------------------------------------------------------------------
// Re-layout a level's f32 output into next level's padded f16 input:
// out[s][t][k] = (s < nseq) ? in[s*T + t][k] : 0
// ------------------------------------------------------------------
__global__ void pack_f32_to_f16(const float* __restrict__ in, _Float16* __restrict__ out,
                                int nseq, int nseq_pad, int T, int K) {
    int idx = blockIdx.x * blockDim.x + threadIdx.x;
    int total = nseq_pad * T * K;
    if (idx >= total) return;
    int k = idx % K;
    int t = (idx / K) % T;
    int s = idx / (K * T);
    float v = (s < nseq) ? in[((size_t)s * T + t) * K + k] : 0.0f;
    out[idx] = (_Float16)v;
}

// ------------------------------------------------------------------
// Fused BiLSTM direction kernel. 16 sequences per block, 8 waves (256 thr).
// Per step: z(16x512) = [x_t | h] @ [Wx;Wh] + b  via v_wmma_f32_16x16x32_f16,
// then gate math in LDS; h kept f16 (A-operand-ready), c kept f32.
// Weights are fragment-swizzled; compiler keeps all 32 B-fragments register-
// resident across the time loop (VGPR-MSB extended file).
// Software pipeline within a step: x A-fragments are loaded FIRST (no
// recurrence dependence), then the serial h-part (ds_load + wmma) executes
// while those global loads are in flight, then x WMMAs consume them.
// ------------------------------------------------------------------
template<int KX>   // padded input feature dim: 128 (sent) or 256 (para/doc)
__global__ __launch_bounds__(256) void lstm_fused(
        const _Float16* __restrict__ x,          // [grid*16][T][KX]
        const unsigned char* __restrict__ mask,  // [nseq][T] (jnp bool)
        const _Float16* __restrict__ wfrag,      // swizzled [(KX+128)/32][NT][32][16]
        const float* __restrict__ bias,          // [512]
        float* __restrict__ out,                 // [grid*16][256]
        int nseq, int T, int reverse, int out_col0) {
    constexpr int KXC = KX / 32;       // x-part K chunks
    __shared__ _Float16 h_lds[16 * 128];
    __shared__ float    c_lds[16 * 128];
    __shared__ float    z_lds[16 * NGATE];

    const int tid = threadIdx.x, lane = tid & 31, wave = tid >> 5;
    const int seqbase = blockIdx.x * 16;
    const size_t xrow = (size_t)T * KX;
    const v16h* wf = (const v16h*)wfrag;

    for (int i = tid; i < 16 * 128; i += 256) { h_lds[i] = (_Float16)0.0f; c_lds[i] = 0.0f; }
    __syncthreads();

    for (int t = 0; t < T; ++t) {
        const int te = reverse ? (T - 1 - t) : t;
        v8f acc[4] = {};

        // ---- (1) issue all x_t A-fragment loads up front (latency hidden
        //          behind the recurrent h-part below) ----
        const _Float16* xb = x + (size_t)seqbase * xrow + (size_t)te * KX;
        v16h axf[KXC];
        #pragma unroll
        for (int kc = 0; kc < KXC; ++kc)
            axf[kc] = load_a16(xb + kc * 32, xrow, lane);

        // ---- (2) recurrent h part of K (the true serial dependency) ----
        #pragma unroll
        for (int kc = 0; kc < 4; ++kc) {
            v16h a = load_a16(h_lds + kc * 32, 128, lane);
            #pragma unroll
            for (int j = 0; j < 4; ++j) {
                v16h b = wf[((size_t)((KXC + kc) * NT + (wave * 4 + j))) * 32 + lane];
                acc[j] = __builtin_amdgcn_wmma_f32_16x16x32_f16(
                            false, a, false, b, (short)0, acc[j], false, false);
            }
        }

        // ---- (3) x_t part of K, consuming the prefetched fragments ----
        #pragma unroll
        for (int kc = 0; kc < KXC; ++kc) {
            #pragma unroll
            for (int j = 0; j < 4; ++j) {
                v16h b = wf[((size_t)(kc * NT + (wave * 4 + j))) * 32 + lane];
                acc[j] = __builtin_amdgcn_wmma_f32_16x16x32_f16(
                            false, axf[kc], false, b, (short)0, acc[j], false, false);
            }
        }

        // ---- bias + stage z to LDS (C layout: n=lane&15, rows (lane>>4)*8 + v) ----
        const int nl = lane & 15, mb = (lane >> 4) * 8;
        #pragma unroll
        for (int j = 0; j < 4; ++j) {
            const int col = wave * 64 + j * 16 + nl;
            const float bv = bias[col];
            #pragma unroll
            for (int v = 0; v < 8; ++v)
                z_lds[(mb + v) * NGATE + col] = acc[j][v] + bv;
        }
        __syncthreads();

        // ---- gates: i|f|g|o = z[:,0:128|128:256|256:384|384:512] ----
        {
            const int m = tid >> 4, n0 = (tid & 15) * 8;
            const int mrow = seqbase + m;
            const bool mk = (mrow < nseq) && mask[(size_t)mrow * T + te];
            #pragma unroll
            for (int q = 0; q < 8; ++q) {
                const int n = n0 + q;
                const float iv = z_lds[m * NGATE + n];
                const float fv = z_lds[m * NGATE + 128 + n];
                const float gv = z_lds[m * NGATE + 256 + n];
                const float ov = z_lds[m * NGATE + 384 + n];
                const float co = c_lds[m * 128 + n];
                const float cn = sigf(fv) * co + sigf(iv) * tanhf(gv);
                const float hn = sigf(ov) * tanhf(cn);
                if (mk) { c_lds[m * 128 + n] = cn; h_lds[m * 128 + n] = (_Float16)hn; }
            }
        }
        __syncthreads();

        // prefetch next timestep's activations (global_prefetch_b8)
        if (wave == 0 && lane < 16 && (t + 1) < T) {
            const int tn = reverse ? (T - 2 - t) : (t + 1);
            __builtin_prefetch(x + (size_t)(seqbase + lane) * xrow + (size_t)tn * KX, 0, 1);
        }
    }

    // write h_last into out[:, out_col0 : out_col0+128]
    const int m = tid >> 4, n0 = (tid & 15) * 8;
    #pragma unroll
    for (int q = 0; q < 8; ++q)
        out[(size_t)(seqbase + m) * 256 + out_col0 + n0 + q] = (float)h_lds[m * 128 + n0 + q];
}

// ------------------------------------------------------------------
// Head: out = sigmoid(tanh(doc @ hidden_w + hidden_b) @ cls_w + cls_b), 2 rows
// ------------------------------------------------------------------
__global__ void head_kernel(const float* __restrict__ doc_out,
                            const float* __restrict__ hw, const float* __restrict__ hb,
                            const float* __restrict__ cw, const float* __restrict__ cb,
                            float* __restrict__ out) {
    __shared__ float h[2 * 256];
    const int tid = threadIdx.x;   // 256 threads, H=256
    for (int b = 0; b < NDOC; ++b) {
        float acc = hb[tid];
        for (int k = 0; k < 256; ++k) acc += doc_out[b * 256 + k] * hw[k * 256 + tid];
        h[b * 256 + tid] = tanhf(acc);
    }
    __syncthreads();
    if (tid < NDOC * 3) {
        const int b = tid / 3, c = tid % 3;
        float acc = cb[c];
        for (int j = 0; j < 256; ++j) acc += h[b * 256 + j] * cw[j * 3 + c];
        out[b * 3 + c] = sigf(acc);
    }
}

// ------------------------------------------------------------------
extern "C" void kernel_launch(void* const* d_in, const int* in_sizes, int n_in,
                              void* d_out, int out_size, void* d_ws, size_t ws_size,
                              hipStream_t stream) {
    const int*           tokens    = (const int*)d_in[0];
    const unsigned char* sent_mask = (const unsigned char*)d_in[1];
    const unsigned char* para_mask = (const unsigned char*)d_in[2];
    const unsigned char* doc_mask  = (const unsigned char*)d_in[3];
    const float* embedding = (const float*)d_in[4];
    const float* w[18];
    for (int i = 0; i < 18; ++i) w[i] = (const float*)d_in[5 + i];
    // w: [0..5]=sent Wx_f,Wh_f,b_f,Wx_b,Wh_b,b_b  [6..11]=para  [12..17]=doc
    const float* hidden_w = (const float*)d_in[23];
    const float* hidden_b = (const float*)d_in[24];
    const float* cls_w    = (const float*)d_in[25];
    const float* cls_b    = (const float*)d_in[26];

    // ---- workspace layout ----
    char* ws = (char*)d_ws;
    size_t off = 0;
    auto alloc = [&](size_t bytes) { char* p = ws + off; off = (off + bytes + 255) & ~(size_t)255; return p; };
    _Float16* emb_f16  = (_Float16*)alloc((size_t)NSENT * SLEN * 128 * 2);   // ~50 MB
    _Float16* wc_sf    = (_Float16*)alloc((size_t)8  * 16384 * 2);
    _Float16* wc_sb    = (_Float16*)alloc((size_t)8  * 16384 * 2);
    _Float16* wc_pf    = (_Float16*)alloc((size_t)12 * 16384 * 2);
    _Float16* wc_pb    = (_Float16*)alloc((size_t)12 * 16384 * 2);
    _Float16* wc_df    = (_Float16*)alloc((size_t)12 * 16384 * 2);
    _Float16* wc_db    = (_Float16*)alloc((size_t)12 * 16384 * 2);
    float*    sent_out = (float*)   alloc((size_t)NSENT * 256 * 4);
    _Float16* para_x   = (_Float16*)alloc((size_t)32 * PLEN * 256 * 2);
    float*    para_out = (float*)   alloc((size_t)32 * 256 * 4);
    _Float16* doc_x    = (_Float16*)alloc((size_t)16 * DLEN * 256 * 2);
    float*    doc_out  = (float*)   alloc((size_t)16 * 256 * 4);

    // 1) embedding gather + f16 convert (pad E 100->128)
    {
        int total = NSENT * SLEN * 128;
        embed_gather<<<(total + 255) / 256, 256, 0, stream>>>(tokens, embedding, emb_f16, NSENT * SLEN);
    }
    // 2) weight prep: [Wx;Wh] swizzled into B-fragment order
    prep_wcat<<<(8  * 16384 + 255) / 256, 256, 0, stream>>>(w[0],  E_DIM, w[1],  wc_sf, 128, 8);
    prep_wcat<<<(8  * 16384 + 255) / 256, 256, 0, stream>>>(w[3],  E_DIM, w[4],  wc_sb, 128, 8);
    prep_wcat<<<(12 * 16384 + 255) / 256, 256, 0, stream>>>(w[6],  256,   w[7],  wc_pf, 256, 12);
    prep_wcat<<<(12 * 16384 + 255) / 256, 256, 0, stream>>>(w[9],  256,   w[10], wc_pb, 256, 12);
    prep_wcat<<<(12 * 16384 + 255) / 256, 256, 0, stream>>>(w[12], 256,   w[13], wc_df, 256, 12);
    prep_wcat<<<(12 * 16384 + 255) / 256, 256, 0, stream>>>(w[15], 256,   w[16], wc_db, 256, 12);

    // 3) sentence BiLSTM: 768 seqs, T=255
    lstm_fused<128><<<NSENT / 16, 256, 0, stream>>>(emb_f16, sent_mask, wc_sf, w[2], sent_out, NSENT, SLEN, 0, 0);
    lstm_fused<128><<<NSENT / 16, 256, 0, stream>>>(emb_f16, sent_mask, wc_sb, w[5], sent_out, NSENT, SLEN, 1, 128);

    // 4) repack -> paragraph input (24 seqs padded to 32, T=32, K=256)
    {
        int total = 32 * PLEN * 256;
        pack_f32_to_f16<<<(total + 255) / 256, 256, 0, stream>>>(sent_out, para_x, NPARA, 32, PLEN, 256);
    }
    lstm_fused<256><<<2, 256, 0, stream>>>(para_x, para_mask, wc_pf, w[8],  para_out, NPARA, PLEN, 0, 0);
    lstm_fused<256><<<2, 256, 0, stream>>>(para_x, para_mask, wc_pb, w[11], para_out, NPARA, PLEN, 1, 128);

    // 5) repack -> doc input (2 seqs padded to 16, T=12, K=256)
    {
        int total = 16 * DLEN * 256;
        pack_f32_to_f16<<<(total + 255) / 256, 256, 0, stream>>>(para_out, doc_x, NDOC, 16, DLEN, 256);
    }
    lstm_fused<256><<<1, 256, 0, stream>>>(doc_x, doc_mask, wc_df, w[14], doc_out, NDOC, DLEN, 0, 0);
    lstm_fused<256><<<1, 256, 0, stream>>>(doc_x, doc_mask, wc_db, w[17], doc_out, NDOC, DLEN, 1, 128);

    // 6) classification head
    head_kernel<<<1, 256, 0, stream>>>(doc_out, hidden_w, hidden_b, cls_w, cls_b, (float*)d_out);
}
